// CausalUnlabeled_2044404433206
// MI455X (gfx1250) — matrix operand order repaired and checked by reading, not compiled, not run on hardware
//
#include <hip/hip_runtime.h>

typedef __attribute__((ext_vector_type(16))) _Float16 v16h;
typedef __attribute__((ext_vector_type(8)))  _Float16 v8h;
typedef __attribute__((ext_vector_type(8)))  float    v8f;

#define BATCH   524288
#define CONT    64
#define NCATE   16
#define VOCAB   1000
#define EM      8
#define LOW     192       // CONT + NCATE*EM
#define RH      32
#define RDIM    32
#define PH      16
#define NHEAD   8

#define NWAVES          4
#define TPB             (NWAVES * 32)
#define ROWS_PER_WAVE   16
#define ROWS_PER_BLOCK  (NWAVES * ROWS_PER_WAVE)

__device__ __forceinline__ v8h load_v8h(const _Float16* p) {
    return *(const v8h*)p;   // all call sites are 16B-aligned by construction
}

__device__ __forceinline__ v16h cat16(v8h lo, v8h hi) {
    return __builtin_shufflevector(lo, hi, 0,1,2,3,4,5,6,7,8,9,10,11,12,13,14,15);
}

__global__ __launch_bounds__(TPB) void mlp_heads_fused(
    const float* __restrict__ x_cont, const int* __restrict__ x_cate,
    const int*  __restrict__ tsel,    const float* __restrict__ emb,
    const float* __restrict__ W1, const float* __restrict__ b1,
    const float* __restrict__ W2, const float* __restrict__ b2,
    const float* __restrict__ W3, const float* __restrict__ b3,
    const float* __restrict__ HW1, const float* __restrict__ Hb1,
    const float* __restrict__ HW2, const float* __restrict__ Hb2,
    float* __restrict__ out)
{
    // ---- LDS (≈58 KB total) ----
    __shared__ __align__(16) _Float16 s_w1t[RH * LOW];     // [n][k]  12 KB
    __shared__ __align__(16) _Float16 s_w2t[RH * RH];      // [n][k]   2 KB
    __shared__ __align__(16) _Float16 s_w3t[RDIM * RH];    // [n][k]   2 KB
    __shared__ float s_b1[RH], s_b2[RH], s_b3[RDIM];
    __shared__ float s_hw1[NHEAD * RDIM * PH];             // 16 KB
    __shared__ float s_hb1[NHEAD * PH];
    __shared__ float s_hw2[NHEAD * PH];
    __shared__ float s_hb2[NHEAD];
    // per-wave staging region: x-tile (16x192 f16) then reused as
    //   h1 (f16, off 0), h2 (f16, off 2048), r (f32, off 4096)
    __shared__ __align__(16) unsigned char s_stage[NWAVES][ROWS_PER_WAVE * LOW * 2];

    const int tid  = threadIdx.x;
    const int w    = tid >> 5;
    const int lane = tid & 31;
    const int half = lane >> 4;   // 0: lanes 0-15, 1: lanes 16-31
    const int l16  = lane & 15;

    // ---- stage weights (block-cooperative, f32 -> f16 transposed) ----
    for (int i = tid; i < RH * LOW; i += TPB) {
        int n = i / LOW, k = i % LOW;
        s_w1t[i] = (_Float16)W1[k * RH + n];
    }
    for (int i = tid; i < RH * RH; i += TPB) {
        int n = i / RH, k = i % RH;
        s_w2t[i] = (_Float16)W2[k * RH + n];
    }
    for (int i = tid; i < RDIM * RH; i += TPB) {
        int n = i / RH, k = i % RH;
        s_w3t[i] = (_Float16)W3[k * RDIM + n];
    }
    for (int i = tid; i < RH; i += TPB) { s_b1[i] = b1[i]; s_b2[i] = b2[i]; s_b3[i] = b3[i]; }
    for (int i = tid; i < NHEAD * RDIM * PH; i += TPB) s_hw1[i] = HW1[i];
    for (int i = tid; i < NHEAD * PH; i += TPB) { s_hb1[i] = Hb1[i]; s_hw2[i] = HW2[i]; }
    if (tid < NHEAD) s_hb2[tid] = Hb2[tid];

    // ---- stage x tile: concat(x_cont, gathered embeddings) as f16 ----
    const long rowBase = (long)blockIdx.x * ROWS_PER_BLOCK + (long)w * ROWS_PER_WAVE;
    _Float16* xs = (_Float16*)s_stage[w];                  // [row][k], k<192

    for (int q = lane; q < ROWS_PER_WAVE * (CONT / 4); q += 32) {
        int row = q >> 4, qi = q & 15;                     // 16 float4 per row
        const float4 v = *(const float4*)(x_cont + (rowBase + row) * CONT + qi * 4);
        _Float16* d = xs + row * LOW + qi * 4;
        d[0] = (_Float16)v.x; d[1] = (_Float16)v.y;
        d[2] = (_Float16)v.z; d[3] = (_Float16)v.w;
    }
#pragma unroll
    for (int p = 0; p < 8; ++p) {                          // 256 (row,cat) gathers / wave
        int task = p * 32 + lane;
        int row = task >> 4, cat = task & 15;
        int idx = x_cate[(rowBase + row) * NCATE + cat];
        const float4* e = (const float4*)(emb + ((long)cat * VOCAB + idx) * EM);
        float4 e0 = e[0], e1 = e[1];
        _Float16* d = xs + row * LOW + CONT + cat * EM;
        d[0] = (_Float16)e0.x; d[1] = (_Float16)e0.y; d[2] = (_Float16)e0.z; d[3] = (_Float16)e0.w;
        d[4] = (_Float16)e1.x; d[5] = (_Float16)e1.y; d[6] = (_Float16)e1.z; d[7] = (_Float16)e1.w;
    }
    __syncthreads();

    // ---- GEMM1: (16x192) @ W1(192x32), 6 K-chunks x 2 N-tiles ----
    v8f acc0 = {0.f,0.f,0.f,0.f,0.f,0.f,0.f,0.f};
    v8f acc1 = {0.f,0.f,0.f,0.f,0.f,0.f,0.f,0.f};
#pragma unroll
    for (int c = 0; c < LOW / 32; ++c) {
        const int kb = c * 32;
        // A fragment: row = l16, one run of 16 f16 at K = kb + half*16
        v16h a = cat16(load_v8h(xs + l16 * LOW + kb + half * 16),
                       load_v8h(xs + l16 * LOW + kb + half * 16 + 8));
        // B fragments: col n, one run of 16 f16 at K = kb + half*16
        v16h bA = cat16(load_v8h(s_w1t + (l16)      * LOW + kb + half * 16),
                        load_v8h(s_w1t + (l16)      * LOW + kb + half * 16 + 8));
        v16h bB = cat16(load_v8h(s_w1t + (16 + l16) * LOW + kb + half * 16),
                        load_v8h(s_w1t + (16 + l16) * LOW + kb + half * 16 + 8));
        acc0 = __builtin_amdgcn_wmma_f32_16x16x32_f16(false, a, false, bA, (short)0, acc0, false, false);
        acc1 = __builtin_amdgcn_wmma_f32_16x16x32_f16(false, a, false, bB, (short)0, acc1, false, false);
    }
    {   // bias + ReLU, store h1 tile (16x32 f16) back to stage region
        _Float16* h1s = (_Float16*)s_stage[w];
        float bv0 = s_b1[l16], bv1 = s_b1[16 + l16];
#pragma unroll
        for (int i = 0; i < 8; ++i) {
            int m = i + half * 8;   // C layout: M = vgpr + half*8, N = l16
            h1s[m * RH + l16]      = (_Float16)fmaxf(acc0[i] + bv0, 0.f);
            h1s[m * RH + 16 + l16] = (_Float16)fmaxf(acc1[i] + bv1, 0.f);
        }
    }
    __syncthreads();

    // ---- small 32x32 GEMMs (K=32: single chunk, 2 N-tiles) ----
    const _Float16* h1s = (const _Float16*)s_stage[w];
    _Float16* h2s = (_Float16*)(s_stage[w] + 2048);
    float*    rts = (float*)(s_stage[w] + 4096);

    auto small_gemm = [&](const _Float16* asrc, const _Float16* bt, v8f& o0, v8f& o1) {
        v16h a = cat16(load_v8h(asrc + l16 * RH + half * 16),
                       load_v8h(asrc + l16 * RH + half * 16 + 8));
        v16h bA = cat16(load_v8h(bt + (l16)      * RH + half * 16),
                        load_v8h(bt + (l16)      * RH + half * 16 + 8));
        v16h bB = cat16(load_v8h(bt + (16 + l16) * RH + half * 16),
                        load_v8h(bt + (16 + l16) * RH + half * 16 + 8));
        o0 = __builtin_amdgcn_wmma_f32_16x16x32_f16(false, a, false, bA, (short)0, o0, false, false);
        o1 = __builtin_amdgcn_wmma_f32_16x16x32_f16(false, a, false, bB, (short)0, o1, false, false);
    };

    // GEMM2: h1 @ W2 + b2, ReLU
    v8f g0 = {0.f,0.f,0.f,0.f,0.f,0.f,0.f,0.f};
    v8f g1 = {0.f,0.f,0.f,0.f,0.f,0.f,0.f,0.f};
    small_gemm(h1s, s_w2t, g0, g1);
    {
        float bv0 = s_b2[l16], bv1 = s_b2[16 + l16];
#pragma unroll
        for (int i = 0; i < 8; ++i) {
            int m = i + half * 8;
            h2s[m * RH + l16]      = (_Float16)fmaxf(g0[i] + bv0, 0.f);
            h2s[m * RH + 16 + l16] = (_Float16)fmaxf(g1[i] + bv1, 0.f);
        }
    }
    __syncthreads();

    // GEMM3: h2 @ W3 + b3 (no ReLU), store r tile as f32
    v8f r0 = {0.f,0.f,0.f,0.f,0.f,0.f,0.f,0.f};
    v8f r1 = {0.f,0.f,0.f,0.f,0.f,0.f,0.f,0.f};
    small_gemm(h2s, s_w3t, r0, r1);
    {
        float bv0 = s_b3[l16], bv1 = s_b3[16 + l16];
#pragma unroll
        for (int i = 0; i < 8; ++i) {
            int m = i + half * 8;
            rts[m * RDIM + l16]      = r0[i] + bv0;
            rts[m * RDIM + 16 + l16] = r1[i] + bv1;
        }
    }
    __syncthreads();

    // ---- head stage: 16 rows x 2 PH-halves across the wave, fp32 VALU ----
    {
        const int  row = l16;
        const long g   = rowBase + row;
        const int  th  = tsel[g];

        float rv[RDIM];
#pragma unroll
        for (int k = 0; k < RDIM; ++k) rv[k] = rts[row * RDIM + k];

        const float* hw1h = s_hw1 + th * RDIM * PH;
        float yp = 0.f;
#pragma unroll
        for (int hh = 0; hh < PH / 2; ++hh) {
            const int h = half * (PH / 2) + hh;
            float s = s_hb1[th * PH + h];
#pragma unroll
            for (int k = 0; k < RDIM; ++k) s = fmaf(rv[k], hw1h[k * PH + h], s);
            s = fmaxf(s, 0.f);
            yp = fmaf(s, s_hw2[th * PH + h], yp);
        }
        float other = __shfl_xor(yp, 16, 32);
        if (half == 0) out[g] = yp + other + s_hb2[th];
    }
}

extern "C" void kernel_launch(void* const* d_in, const int* in_sizes, int n_in,
                              void* d_out, int out_size, void* d_ws, size_t ws_size,
                              hipStream_t stream) {
    (void)in_sizes; (void)n_in; (void)out_size; (void)d_ws; (void)ws_size;
    const float* x_cont = (const float*)d_in[0];
    const int*   x_cate = (const int*)  d_in[1];
    const int*   tsel   = (const int*)  d_in[2];
    const float* emb    = (const float*)d_in[3];
    const float* W1 = (const float*)d_in[4];  const float* b1 = (const float*)d_in[5];
    const float* W2 = (const float*)d_in[6];  const float* b2 = (const float*)d_in[7];
    const float* W3 = (const float*)d_in[8];  const float* b3 = (const float*)d_in[9];
    const float* HW1 = (const float*)d_in[10]; const float* Hb1 = (const float*)d_in[11];
    const float* HW2 = (const float*)d_in[12]; const float* Hb2 = (const float*)d_in[13];
    float* out = (float*)d_out;

    const int blocks = BATCH / ROWS_PER_BLOCK;   // 8192
    hipLaunchKernelGGL(mlp_heads_fused, dim3(blocks), dim3(TPB), 0, stream,
                       x_cont, x_cate, tsel, emb,
                       W1, b1, W2, b2, W3, b3, HW1, Hb1, HW2, Hb2, out);
}
